// AdditiveRelationalGraphConvolution_5909875000114
// MI455X (gfx1250) — compile-verified, matrix operand order checked by reading, not run on hardware
//
#include <hip/hip_runtime.h>
#include <hip/hip_bf16.h>

typedef __attribute__((ext_vector_type(16))) _Float16 v16h;
typedef __attribute__((ext_vector_type(8)))  _Float16 v8h;
typedef __attribute__((ext_vector_type(8)))  float    v8f;

#define NUM_NODES 100000
#define NUM_REL   238      // NUM_REL_P1
#define BATCH     32768
#define SMP       16
#define DIN       256
#define DOUT      256

#define A_STRIDE  264      // f16 elems per LDS A row (16B pad -> conflict-free b128 reads)
#define R_STRIDE  260      // f32 elems per LDS R row (pad to dodge half-wave conflicts)

// ---- prep: weight f32 [DOUT][DIN] -> f16 [DOUT][DIN] (K-contiguous = WMMA B layout) ----
__global__ void prep_weight_f16(const float* __restrict__ w, _Float16* __restrict__ wf16) {
    int idx = blockIdx.x * 256 + threadIdx.x;   // grid sized exactly DOUT*DIN/256
    wf16[idx] = (_Float16)w[idx];
}

// ---- prep: relation_weight [DOUT][NUM_REL] -> relT [NUM_REL][DOUT] (row-major gather) ----
__global__ void prep_rel_transpose(const float* __restrict__ relw, float* __restrict__ relT) {
    int idx = blockIdx.x * 256 + threadIdx.x;   // grid = NUM_REL blocks, exact
    int r = idx >> 8;            // / DOUT
    int d = idx & 255;           // % DOUT
    relT[idx] = relw[d * NUM_REL + r];
}

// ---- fused: gather+mean -> LDS, WMMA GEMM vs weight_f16, + rel mean, ReLU ----
__global__ __launch_bounds__(256)
void rgcn_fused(const int*      __restrict__ neighbors,
                const int*      __restrict__ relations,
                const float*    __restrict__ features,
                const _Float16* __restrict__ wf16,
                const float*    __restrict__ relT,
                float*          __restrict__ out) {
    __shared__ __align__(16) _Float16 sA[16 * A_STRIDE];  // 8448 B
    __shared__ __align__(16) float    sR[16 * R_STRIDE];  // 16640 B

    const int lane  = threadIdx.x & 31;
    const int wave  = threadIdx.x >> 5;
    const int rbase = blockIdx.x << 4;     // 16 batch rows per block
    const int col   = lane << 3;           // this lane's 8-wide column slice

    // ---------------- aggregation phase ----------------
    for (int rr = 0; rr < 2; ++rr) {
        const int row = wave * 2 + rr;
        const int gb  = (rbase + row) * SMP;

        float a0=0.f,a1=0.f,a2=0.f,a3=0.f,a4=0.f,a5=0.f,a6=0.f,a7=0.f;
        #pragma unroll 4
        for (int s = 0; s < SMP; ++s) {
            const int nbr = neighbors[gb + s];
            const float4* fp = (const float4*)(features + (size_t)nbr * DIN + col);
            float4 x = fp[0], y = fp[1];
            a0 += x.x; a1 += x.y; a2 += x.z; a3 += x.w;
            a4 += y.x; a5 += y.y; a6 += y.z; a7 += y.w;
        }
        const float sc = 1.0f / 16.0f;
        v8h h;
        h[0]=(_Float16)(a0*sc); h[1]=(_Float16)(a1*sc);
        h[2]=(_Float16)(a2*sc); h[3]=(_Float16)(a3*sc);
        h[4]=(_Float16)(a4*sc); h[5]=(_Float16)(a5*sc);
        h[6]=(_Float16)(a6*sc); h[7]=(_Float16)(a7*sc);
        *(v8h*)(sA + row * A_STRIDE + col) = h;

        float r0=0.f,r1=0.f,r2=0.f,r3=0.f,r4=0.f,r5=0.f,r6=0.f,r7=0.f;
        #pragma unroll 4
        for (int s = 0; s < SMP; ++s) {
            const int rel = relations[gb + s];
            const float4* rp = (const float4*)(relT + (size_t)rel * DOUT + col);
            float4 x = rp[0], y = rp[1];
            r0 += x.x; r1 += x.y; r2 += x.z; r3 += x.w;
            r4 += y.x; r5 += y.y; r6 += y.z; r7 += y.w;
        }
        float* dst = sR + row * R_STRIDE + col;
        float4 rv0 = make_float4(r0*sc, r1*sc, r2*sc, r3*sc);
        float4 rv1 = make_float4(r4*sc, r5*sc, r6*sc, r7*sc);
        ((float4*)dst)[0] = rv0;
        ((float4*)dst)[1] = rv1;
    }
    __syncthreads();

    // ---------------- WMMA phase ----------------
    // A fragment (16x32 f16): lane -> row = lane&15, khalf = (lane>>4)*8;
    // per k-step needs f16 K-slices [kb, kb+8) and [kb+16, kb+24).
    const int nrow  = lane & 15;
    const int khalf = (lane >> 4) << 3;
    const int n0 = wave * 2;
    const int n1 = n0 + 1;

    const _Float16* abase  = sA  + nrow * A_STRIDE;
    const _Float16* bbase0 = wf16 + (size_t)(n0 * 16 + nrow) * DIN;
    const _Float16* bbase1 = wf16 + (size_t)(n1 * 16 + nrow) * DIN;

    v8f acc0 = {}; v8f acc1 = {};
    #pragma unroll
    for (int ks = 0; ks < 8; ++ks) {
        const int kb = ks * 32 + khalf;

        v8h alo = *(const v8h*)(abase + kb);
        v8h ahi = *(const v8h*)(abase + kb + 16);
        v16h a = __builtin_shufflevector(alo, ahi,
                 0,1,2,3,4,5,6,7,8,9,10,11,12,13,14,15);

        v8h b0lo = *(const v8h*)(bbase0 + kb);
        v8h b0hi = *(const v8h*)(bbase0 + kb + 16);
        v16h b0 = __builtin_shufflevector(b0lo, b0hi,
                  0,1,2,3,4,5,6,7,8,9,10,11,12,13,14,15);
        acc0 = __builtin_amdgcn_wmma_f32_16x16x32_f16(
                   false, a, false, b0, (short)0, acc0, false, false);

        v8h b1lo = *(const v8h*)(bbase1 + kb);
        v8h b1hi = *(const v8h*)(bbase1 + kb + 16);
        v16h b1 = __builtin_shufflevector(b1lo, b1hi,
                  0,1,2,3,4,5,6,7,8,9,10,11,12,13,14,15);
        acc1 = __builtin_amdgcn_wmma_f32_16x16x32_f16(
                   false, a, false, b1, (short)0, acc1, false, false);
    }

    // ---------------- epilogue ----------------
    // C layout: VGPR r, lane L -> M = r + 8*(L>>4), N = L&15
    const int mhalf = (lane >> 4) << 3;
    const int ncol  = lane & 15;
    #pragma unroll
    for (int r = 0; r < 8; ++r) {
        const int m = r + mhalf;
        float v0 = acc0[r] + sR[m * R_STRIDE + n0 * 16 + ncol];
        v0 = fmaxf(v0, 0.0f);
        out[(size_t)(rbase + m) * DOUT + n0 * 16 + ncol] = v0;

        float v1 = acc1[r] + sR[m * R_STRIDE + n1 * 16 + ncol];
        v1 = fmaxf(v1, 0.0f);
        out[(size_t)(rbase + m) * DOUT + n1 * 16 + ncol] = v1;
    }
}

extern "C" void kernel_launch(void* const* d_in, const int* in_sizes, int n_in,
                              void* d_out, int out_size, void* d_ws, size_t ws_size,
                              hipStream_t stream) {
    const int*   neighbors = (const int*)d_in[0];
    const int*   relations = (const int*)d_in[1];
    const float* features  = (const float*)d_in[2];
    const float* weight    = (const float*)d_in[3];
    const float* relw      = (const float*)d_in[4];
    float*       out       = (float*)d_out;

    // workspace layout: relT [238][256] f32, then weight f16 [256][256]
    float*    relT = (float*)d_ws;
    _Float16* wf16 = (_Float16*)((char*)d_ws + (size_t)NUM_REL * DOUT * sizeof(float));

    prep_weight_f16   <<<(DOUT * DIN) / 256, 256, 0, stream>>>(weight, wf16);
    prep_rel_transpose<<<NUM_REL,            256, 0, stream>>>(relw, relT);
    rgcn_fused        <<<BATCH / 16,         256, 0, stream>>>(neighbors, relations,
                                                               features, wf16, relT, out);
}
